// EnhancedRWKVBlock_37864431681707
// MI455X (gfx1250) — matrix operand order repaired and checked by reading, not compiled, not run on hardware
//
#include <hip/hip_runtime.h>
#include <stdint.h>

#define B_  4
#define T_  2048
#define H_  2048
#define D_  4
#define FF_ 8192
#define M_  (B_*T_)

typedef __attribute__((ext_vector_type(16))) __bf16        v16bf;
typedef __attribute__((ext_vector_type(8)))  float         v8f;
typedef __attribute__((ext_vector_type(4)))  unsigned int  u4;

__device__ __forceinline__ unsigned short f2bf(float f) {
    unsigned int u = __builtin_bit_cast(unsigned int, f);
    u += 0x7FFFu + ((u >> 16) & 1u);           // round-to-nearest-even
    return (unsigned short)(u >> 16);
}
__device__ __forceinline__ float bf2f(unsigned short h) {
    unsigned int u = ((unsigned int)h) << 16;
    return __builtin_bit_cast(float, u);
}
__device__ __forceinline__ float sigm(float x) { return 1.0f / (1.0f + __expf(-x)); }

// ---------------------------------------------------------------------------
// CDNA5 async global->LDS copy (VGLOBAL GLOBAL_LOAD_ASYNC_TO_LDS_B128, ASYNCcnt)
// ---------------------------------------------------------------------------
__device__ __forceinline__ void cp_async_b128(unsigned lds_addr, unsigned long long gaddr) {
    asm volatile("global_load_async_to_lds_b128 %0, %1, off"
                 :: "v"(lds_addr), "v"(gaddr) : "memory");
}
template <int N>
__device__ __forceinline__ void wait_async() {
#if __has_builtin(__builtin_amdgcn_s_wait_asynccnt)
    __builtin_amdgcn_s_wait_asynccnt(N);
#else
    asm volatile("s_wait_asynccnt %0" :: "i"(N) : "memory");
#endif
}

// ---------------------------------------------------------------------------
// Weight transpose + f32->bf16 convert.  dst is [N][K] bf16; src[k*srcLd + n].
// ---------------------------------------------------------------------------
__global__ __launch_bounds__(256) void transpose_cvt(
    const float* __restrict__ src, unsigned short* __restrict__ dst,
    int K, int N, int srcLd)
{
    size_t idx = (size_t)blockIdx.x * 256 + threadIdx.x;
    size_t n = idx / (size_t)K;
    size_t k = idx % (size_t)K;
    dst[idx] = f2bf(src[k * (size_t)srcLd + n]);
}

// sdec[b,d,h] = att_state[b,d,h] * exp(-exp(td_multi[d,h]))   (B*D*H = 32768)
__global__ __launch_bounds__(256) void sdec_kernel(
    const float* __restrict__ att_state, const float* __restrict__ td,
    float* __restrict__ sdec)
{
    int idx = blockIdx.x * 256 + threadIdx.x;
    sdec[idx] = att_state[idx] * __expf(-__expf(td[idx & (D_ * H_ - 1)]));
}

// ---------------------------------------------------------------------------
// LayerNorm over H=2048, one row per block, bf16 output.
// ---------------------------------------------------------------------------
__global__ __launch_bounds__(256) void ln_kernel(
    const float* __restrict__ x, const float* __restrict__ s,
    const float* __restrict__ b, unsigned short* __restrict__ out)
{
    const int tid = threadIdx.x;
    const size_t row = blockIdx.x;
    const float* xr = x + row * H_;
    float sum = 0.f, sq = 0.f;
    for (int i = tid; i < H_; i += 256) { float v = xr[i]; sum += v; sq += v * v; }
    __shared__ float r1[256], r2[256];
    r1[tid] = sum; r2[tid] = sq; __syncthreads();
    for (int st = 128; st > 0; st >>= 1) {
        if (tid < st) { r1[tid] += r1[tid + st]; r2[tid] += r2[tid + st]; }
        __syncthreads();
    }
    const float mean = r1[0] * (1.0f / H_);
    const float var  = r2[0] * (1.0f / H_) - mean * mean;
    const float rs   = rsqrtf(var + 1e-5f);
    for (int i = tid; i < H_; i += 256)
        out[row * H_ + i] = f2bf((xr[i] - mean) * rs * s[i] + b[i]);
}

// ---------------------------------------------------------------------------
// Fused: lw = softmax(h @ lvl_w + lvl_b) over D=4; g = sigmoid(r)*(Σ lw*sdec + k*v)
// y1 packs [v | k | r_pre] along N (cols 0..2047 | 2048.. | 4096..).
// ---------------------------------------------------------------------------
__global__ __launch_bounds__(256) void attn_gate_kernel(
    const unsigned short* __restrict__ h, const unsigned short* __restrict__ y1,
    const float* __restrict__ sdec, const float* __restrict__ lvl_w,
    const float* __restrict__ lvl_b, unsigned short* __restrict__ g)
{
    const int tid = threadIdx.x;
    const size_t row = blockIdx.x;
    const int bidx = (int)(row / T_);
    const unsigned short* hr = h + row * H_;

    float l[D_] = {0.f, 0.f, 0.f, 0.f};
    for (int i = tid; i < H_; i += 256) {
        float hv = bf2f(hr[i]);
        l[0] += hv * lvl_w[i * D_ + 0];
        l[1] += hv * lvl_w[i * D_ + 1];
        l[2] += hv * lvl_w[i * D_ + 2];
        l[3] += hv * lvl_w[i * D_ + 3];
    }
    __shared__ float red[256];
    __shared__ float lg[D_];
    __shared__ float lw[D_];
    for (int d = 0; d < D_; ++d) {
        red[tid] = l[d]; __syncthreads();
        for (int st = 128; st > 0; st >>= 1) {
            if (tid < st) red[tid] += red[tid + st];
            __syncthreads();
        }
        if (tid == 0) lg[d] = red[0] + lvl_b[d];
        __syncthreads();
    }
    if (tid == 0) {
        float m = lg[0];
        for (int d = 1; d < D_; ++d) m = fmaxf(m, lg[d]);
        float e[D_], sum = 0.f;
        for (int d = 0; d < D_; ++d) { e[d] = __expf(lg[d] - m); sum += e[d]; }
        float inv = 1.0f / sum;
        for (int d = 0; d < D_; ++d) lw[d] = e[d] * inv;
    }
    __syncthreads();

    const unsigned short* yr = y1 + row * (size_t)(3 * H_);
    const float* sd = sdec + (size_t)bidx * D_ * H_;
    const float w0 = lw[0], w1 = lw[1], w2 = lw[2], w3 = lw[3];
    for (int i = tid; i < H_; i += 256) {
        float vv = bf2f(yr[i]);
        float kk = bf2f(yr[H_ + i]);
        float rp = bf2f(yr[2 * H_ + i]);
        float a = w0 * sd[i] + w1 * sd[H_ + i] + w2 * sd[2 * H_ + i] + w3 * sd[3 * H_ + i];
        g[row * H_ + i] = f2bf(sigm(rp) * (a + kk * vv));
    }
}

// km = h2*tmk + shift(h2,cm_state)*(1-tmk)
__global__ __launch_bounds__(256) void mix_kernel(
    const unsigned short* __restrict__ h2, const float* __restrict__ cm_state,
    const float* __restrict__ tmk, unsigned short* __restrict__ km)
{
    const int tid = threadIdx.x;
    const size_t row = blockIdx.x;
    const int t = (int)(row % T_);
    const int bidx = (int)(row / T_);
    for (int i = tid; i < H_; i += 256) {
        float cur  = bf2f(h2[row * H_ + i]);
        float prev = (t == 0) ? cm_state[(size_t)bidx * H_ + i]
                              : bf2f(h2[(row - 1) * H_ + i]);
        float tk = tmk[i];
        km[row * H_ + i] = f2bf(cur * tk + prev * (1.0f - tk));
    }
}

// d_out = x1 + val*sigmoid(gate); y45 packs [val | gate] along N (4096 cols)
__global__ __launch_bounds__(256) void final_kernel(
    const unsigned short* __restrict__ y45, float* __restrict__ out)
{
    const int tid = threadIdx.x;
    const size_t row = blockIdx.x;
    const unsigned short* yr = y45 + row * (size_t)(2 * H_);
    for (int i = tid; i < H_; i += 256) {
        float v  = bf2f(yr[i]);
        float gt = bf2f(yr[H_ + i]);
        out[row * H_ + i] += v * sigm(gt);
    }
}

// ---------------------------------------------------------------------------
// bf16 WMMA GEMM: C[M,N] = A[M,K] @ B[K,N], Bt is B^T laid out [N][K] bf16.
// Block tile 128(M) x 256(N), BK=32, 8 waves, wave tile 64x64 (16 WMMA/step).
// Double-buffered LDS filled by global_load_async_to_lds_b128 (ASYNCcnt).
// EPI: 0 = store bf16; 1 = relu^2 then bf16; 2 = +resid then f32.
// ---------------------------------------------------------------------------
template <int EPI>
__global__ __launch_bounds__(256) void gemm_bf16_wmma(
    const unsigned short* __restrict__ A, const unsigned short* __restrict__ Bt,
    void* __restrict__ C, const float* __restrict__ resid,
    int M, int N, int K)
{
    __shared__ u4 As4[2 * 512];    // 2 x (128 rows x 32 bf16)  = 16 KB
    __shared__ u4 Bs4[2 * 1024];   // 2 x (256 rows x 32 bf16)  = 32 KB

    const int tid   = threadIdx.x;
    const int lane  = tid & 31;
    const int wave  = tid >> 5;
    const int waveM = wave >> 2;          // 0..1 -> 64-row strip
    const int waveN = wave & 3;           // 0..3 -> 64-col strip
    const int m0 = blockIdx.y * 128;
    const int n0 = blockIdx.x * 256;

    const int lr = tid >> 1;              // 0..127 : A-tile row loaded by this thread
    const int hs = tid & 1;               // which 16-element half of the 32-wide row
    const int l15 = lane & 15;
    const int lh  = lane >> 4;

    const unsigned ldsA = (unsigned)(size_t)(void*)&As4[0];
    const unsigned ldsB = (unsigned)(size_t)(void*)&Bs4[0];

    v8f acc[4][4] = {};

    const int nk = K >> 5;

    // issue one K-tile's async copies: 2 b128 for A + 4 b128 for B per thread
    auto issue_tile = [&](int k0, int buf) {
        unsigned long long ga =
            (unsigned long long)(size_t)(A + (size_t)(m0 + lr) * K + k0 + 16 * hs);
        unsigned la = ldsA + (unsigned)(buf * 512 + lr * 4 + 2 * hs) * 16u;
        cp_async_b128(la,        ga);
        cp_async_b128(la + 16u,  ga + 16ull);
        unsigned long long gb =
            (unsigned long long)(size_t)(Bt + (size_t)(n0 + tid) * K + k0);
        unsigned lb = ldsB + (unsigned)(buf * 1024 + tid * 4) * 16u;
        cp_async_b128(lb,        gb);
        cp_async_b128(lb + 16u,  gb + 16ull);
        cp_async_b128(lb + 32u,  gb + 32ull);
        cp_async_b128(lb + 48u,  gb + 48ull);
    };

    issue_tile(0, 0);

    for (int kt = 0; kt < nk; ++kt) {
        const int buf = kt & 1;
        if (kt + 1 < nk) {
            issue_tile((kt + 1) << 5, buf ^ 1);
            wait_async<6>();              // oldest 6 (this tile) complete, in-order
        } else {
            wait_async<0>();
        }
        __syncthreads();                  // all waves' copies for this tile landed

        // B fragments: lane holds col (lane&15), 16 contiguous K at 16*(lane>>4)
        union { v16bf v; u4 q[2]; } bfr[4];
#pragma unroll
        for (int ni = 0; ni < 4; ++ni) {
            int bn = waveN * 64 + ni * 16 + l15;
            bfr[ni].q[0] = Bs4[buf * 1024 + bn * 4 + 2 * lh];
            bfr[ni].q[1] = Bs4[buf * 1024 + bn * 4 + 2 * lh + 1];
        }
#pragma unroll
        for (int mi = 0; mi < 4; ++mi) {
            // A fragment: lane holds row (lane&15), 16 contiguous K at 16*(lane>>4)
            int am = waveM * 64 + mi * 16 + l15;
            union { v16bf v; u4 q[2]; } afr;
            afr.q[0] = As4[buf * 512 + am * 4 + 2 * lh];
            afr.q[1] = As4[buf * 512 + am * 4 + 2 * lh + 1];
#pragma unroll
            for (int ni = 0; ni < 4; ++ni) {
                acc[mi][ni] = __builtin_amdgcn_wmma_f32_16x16x32_bf16(
                    false, afr.v, false, bfr[ni].v,
                    (short)0, acc[mi][ni], false, false);
            }
        }
        __syncthreads();                  // safe to overwrite buf next iteration
    }

    // Epilogue.  C layout: m = i + 8*(lane>>4), n = lane&15 per 16x16 tile.
#pragma unroll
    for (int mi = 0; mi < 4; ++mi) {
#pragma unroll
        for (int ni = 0; ni < 4; ++ni) {
            int rbase = m0 + waveM * 64 + mi * 16 + 8 * lh;
            int col   = n0 + waveN * 64 + ni * 16 + l15;
#pragma unroll
            for (int i = 0; i < 8; ++i) {
                size_t idx = (size_t)(rbase + i) * N + col;
                float v = acc[mi][ni][i];
                if (EPI == 1) { v = v > 0.f ? v : 0.f; v = v * v; }
                if (EPI == 2) {
                    ((float*)C)[idx] = v + resid[idx];
                } else {
                    ((unsigned short*)C)[idx] = f2bf(v);
                }
            }
        }
    }
}

// ---------------------------------------------------------------------------
extern "C" void kernel_launch(void* const* d_in, const int* in_sizes, int n_in,
                              void* d_out, int out_size, void* d_ws, size_t ws_size,
                              hipStream_t stream)
{
    (void)in_sizes; (void)n_in; (void)out_size; (void)ws_size;
    const float* x         = (const float*)d_in[0];
    const float* att_state = (const float*)d_in[1];
    const float* cm_state  = (const float*)d_in[2];
    const float* ln1_s     = (const float*)d_in[3];
    const float* ln1_b     = (const float*)d_in[4];
    const float* ln2_s     = (const float*)d_in[5];
    const float* ln2_b     = (const float*)d_in[6];
    const float* td        = (const float*)d_in[7];
    const float* lvl_w     = (const float*)d_in[8];
    const float* lvl_b     = (const float*)d_in[9];
    const float* Wv        = (const float*)d_in[10];
    const float* Wk        = (const float*)d_in[11];
    const float* Wr        = (const float*)d_in[12];
    const float* Wo        = (const float*)d_in[13];
    const float* tmk       = (const float*)d_in[14];
    const float* Wkey      = (const float*)d_in[15];
    const float* Wval      = (const float*)d_in[16];
    const float* Wgate     = (const float*)d_in[17];
    float* out = (float*)d_out;

    char* ws = (char*)d_ws;
    size_t off = 0;
    auto arena = [&](size_t bytes) -> char* {
        char* p = ws + off;
        off = (off + bytes + 255) & ~(size_t)255;
        return p;
    };
    unsigned short* wcat1  = (unsigned short*)arena((size_t)3 * H_ * H_ * 2);  // [Wv|Wk|Wr]^T
    unsigned short* wo_t   = (unsigned short*)arena((size_t)H_ * H_ * 2);
    unsigned short* wkey_t = (unsigned short*)arena((size_t)FF_ * H_ * 2);
    unsigned short* wvg_t  = (unsigned short*)arena((size_t)2 * H_ * FF_ * 2); // [Wval|Wgate]^T
    unsigned short* hbuf   = (unsigned short*)arena((size_t)M_ * H_ * 2);      // h, later km
    unsigned short* y1     = (unsigned short*)arena((size_t)M_ * 3 * H_ * 2);  // v|k|r, later val|gate
    unsigned short* gbuf   = (unsigned short*)arena((size_t)M_ * H_ * 2);      // g, later h2
    unsigned short* kkbuf  = (unsigned short*)arena((size_t)M_ * FF_ * 2);
    float*          sdec   = (float*)arena((size_t)B_ * D_ * H_ * 4);

    dim3 blk(256);

    // ---- weights -> bf16 transposed [N][K] ----
    transpose_cvt<<<(H_ * H_) / 256, blk, 0, stream>>>(Wv, wcat1,                        H_, H_, H_);
    transpose_cvt<<<(H_ * H_) / 256, blk, 0, stream>>>(Wk, wcat1 + (size_t)H_ * H_,      H_, H_, H_);
    transpose_cvt<<<(H_ * H_) / 256, blk, 0, stream>>>(Wr, wcat1 + (size_t)2 * H_ * H_,  H_, H_, H_);
    transpose_cvt<<<(H_ * H_) / 256, blk, 0, stream>>>(Wo, wo_t,                         H_, H_, H_);
    transpose_cvt<<<(H_ * FF_) / 256, blk, 0, stream>>>(Wkey,  wkey_t,                    H_, FF_, FF_);
    transpose_cvt<<<(FF_ * H_) / 256, blk, 0, stream>>>(Wval,  wvg_t,                     FF_, H_, H_);
    transpose_cvt<<<(FF_ * H_) / 256, blk, 0, stream>>>(Wgate, wvg_t + (size_t)H_ * FF_,  FF_, H_, H_);

    sdec_kernel<<<(B_ * D_ * H_) / 256, blk, 0, stream>>>(att_state, td, sdec);

    // ---- attention path ----
    ln_kernel<<<M_, blk, 0, stream>>>(x, ln1_s, ln1_b, hbuf);

    dim3 g1((3 * H_) / 256, M_ / 128);
    gemm_bf16_wmma<0><<<g1, blk, 0, stream>>>(hbuf, wcat1, y1, nullptr, M_, 3 * H_, H_);

    attn_gate_kernel<<<M_, blk, 0, stream>>>(hbuf, y1, sdec, lvl_w, lvl_b, gbuf);

    dim3 g2(H_ / 256, M_ / 128);
    gemm_bf16_wmma<2><<<g2, blk, 0, stream>>>(gbuf, wo_t, out, x, M_, H_, H_);   // x1 -> d_out

    // ---- channel-mix path ----
    ln_kernel<<<M_, blk, 0, stream>>>(out, ln2_s, ln2_b, gbuf);                  // h2
    mix_kernel<<<M_, blk, 0, stream>>>(gbuf, cm_state, tmk, hbuf);               // km

    dim3 g3(FF_ / 256, M_ / 128);
    gemm_bf16_wmma<1><<<g3, blk, 0, stream>>>(hbuf, wkey_t, kkbuf, nullptr, M_, FF_, H_); // relu^2

    dim3 g4((2 * H_) / 256, M_ / 128);
    gemm_bf16_wmma<0><<<g4, blk, 0, stream>>>(kkbuf, wvg_t, y1, nullptr, M_, 2 * H_, FF_);

    final_kernel<<<M_, blk, 0, stream>>>(y1, out);
}
